// tDBN_bv_2_55044300866043
// MI455X (gfx1250) — compile-verified, matrix operand order checked by reading, not run on hardware
//
#include <hip/hip_runtime.h>

// ---------------------------------------------------------------------------
// CDNA5 / gfx1250 implementation of the tDBN sparse-3D-UNet forward pass.
// Heavy 3x3x3 convs run as implicit-GEMM on v_wmma_f32_16x16x32_bf16 with:
//   - mask-driven zero-tile fast path (grid ~2% occupied)
//   - fragment-major pre-swizzled bf16 weights (2x global_load_b128 per lane)
//   - LDS patch with padded pitch (bank-conflict-free ds_load_b128)
//   - TDM (tensor_load_to_lds + s_wait_tensorcnt) staging for interior tiles
// ---------------------------------------------------------------------------

typedef __attribute__((ext_vector_type(16))) __bf16 v16bf;
typedef __attribute__((ext_vector_type(8)))  float  v8f;

union WFrag { v16bf v; unsigned short u[16]; uint4 q[2]; };

#if defined(__HIP_DEVICE_COMPILE__) && __has_builtin(__builtin_amdgcn_tensor_load_to_lds) && \
    __has_builtin(__builtin_amdgcn_s_wait_tensorcnt)
#define USE_TDM 1
typedef __attribute__((ext_vector_type(4))) unsigned int v4u_;
typedef __attribute__((ext_vector_type(8))) int v8i_;
typedef __attribute__((ext_vector_type(4))) int v4i_;
#else
#define USE_TDM 0
#endif

__device__ __forceinline__ unsigned short f2bf(float f) {
    unsigned int u = __float_as_uint(f);
    unsigned int r = u + 0x7FFFu + ((u >> 16) & 1u);   // round-to-nearest-even
    return (unsigned short)(r >> 16);
}
__device__ __forceinline__ float bf2f(unsigned short b) {
    return __uint_as_float(((unsigned int)b) << 16);
}

// ------------------------------ elementwise --------------------------------

__global__ void k_zero_f32(float* p, size_t n) {
    for (size_t i = (size_t)blockIdx.x * blockDim.x + threadIdx.x; i < n;
         i += (size_t)gridDim.x * blockDim.x) p[i] = 0.f;
}
__global__ void k_zero_u16(unsigned short* p, size_t n) {
    for (size_t i = (size_t)blockIdx.x * blockDim.x + threadIdx.x; i < n;
         i += (size_t)gridDim.x * blockDim.x) p[i] = 0;
}
__global__ void k_zero_u8(unsigned char* p, size_t n) {
    for (size_t i = (size_t)blockIdx.x * blockDim.x + threadIdx.x; i < n;
         i += (size_t)gridDim.x * blockDim.x) p[i] = 0;
}
__global__ void k_f32_to_bf16(const float* __restrict__ s, unsigned short* __restrict__ d, size_t n) {
    for (size_t i = (size_t)blockIdx.x * blockDim.x + threadIdx.x; i < n;
         i += (size_t)gridDim.x * blockDim.x) d[i] = f2bf(s[i]);
}
__global__ void k_add2_f32(float* __restrict__ dst, const float* __restrict__ a,
                           const float* __restrict__ b, size_t n) {
    for (size_t i = (size_t)blockIdx.x * blockDim.x + threadIdx.x; i < n;
         i += (size_t)gridDim.x * blockDim.x) dst[i] = a[i] + b[i];
}
__global__ void k_concat(const float* __restrict__ a, const float* __restrict__ b,
                         float* __restrict__ dst, size_t nvox, int Ca, int Cb) {
    int C = Ca + Cb;
    size_t tot = nvox * (size_t)C;
    for (size_t i = (size_t)blockIdx.x * blockDim.x + threadIdx.x; i < tot;
         i += (size_t)gridDim.x * blockDim.x) {
        int c = (int)(i % C); size_t v = i / C;
        dst[i] = (c < Ca) ? a[v * Ca + c] : b[v * Cb + (c - Ca)];
    }
}

// weight swizzle: [KD*KH*KW, Cin, Cout] f32  ->  fragment-major bf16
// dst[((tap*Q + kq)*S + cs)*32 + lane][16 halves], Q=Cin/32, S=Cout/16
__global__ void k_swizzle_w(const float* __restrict__ src, unsigned short* __restrict__ dst,
                            int taps, int Cin, int Cout) {
    size_t tot = (size_t)taps * Cin * Cout;
    int Q = Cin >> 5, S = Cout >> 4;
    for (size_t i = (size_t)blockIdx.x * blockDim.x + threadIdx.x; i < tot;
         i += (size_t)gridDim.x * blockDim.x) {
        int j = (int)(i & 15); size_t r = i >> 4;
        int lane = (int)(r & 31); r >>= 5;
        int cs = (int)(r % S); r /= S;
        int kq = (int)(r % Q); int t = (int)(r / Q);
        int h = j & 1, vv = j >> 1, nh = lane >> 4;
        int kk = (vv < 4 ? nh * 8 + 2 * vv : 16 + nh * 8 + 2 * (vv - 4)) + h;
        int cin = kq * 32 + kk;
        int co  = cs * 16 + (lane & 15);
        dst[i] = f2bf(src[((size_t)t * Cin + cin) * Cout + co]);
    }
}

// ------------------------------ scatter / masks ----------------------------

__global__ void k_scatter(const float* __restrict__ vf, const int* __restrict__ coors,
                          unsigned short* __restrict__ grid, unsigned char* __restrict__ mask,
                          int n, int G) {
    for (size_t i = (size_t)blockIdx.x * blockDim.x + threadIdx.x; i < (size_t)n;
         i += (size_t)gridDim.x * blockDim.x) {
        int z = coors[i * 4 + 1], y = coors[i * 4 + 2], x = coors[i * 4 + 3];
        size_t idx = ((size_t)z * G + y) * G + x;
        grid[idx] = f2bf(vf[i]);
        mask[idx] = 1;
    }
}

__global__ void k_down_mask(const unsigned char* __restrict__ m, unsigned char* __restrict__ mo,
                            int Di, int Do) {
    size_t tot = (size_t)Do * Do * Do;
    for (size_t i = (size_t)blockIdx.x * blockDim.x + threadIdx.x; i < tot;
         i += (size_t)gridDim.x * blockDim.x) {
        int x = (int)(i % Do); int y = (int)((i / Do) % Do); int z = (int)(i / ((size_t)Do * Do));
        unsigned char v = 0;
        for (int dz = 0; dz < 3; ++dz)
            for (int dy = 0; dy < 3; ++dy)
                for (int dx = 0; dx < 3; ++dx) {
                    int iz = z * 2 + dz, iy = y * 2 + dy, ix = x * 2 + dx;
                    if (iz < Di && iy < Di && ix < Di) v |= m[((size_t)iz * Di + iy) * Di + ix];
                }
        mo[i] = v;
    }
}

__global__ void k_reduce_mask_d(const unsigned char* __restrict__ m, unsigned char* __restrict__ mo,
                                int D, int P, int dk) {
    int Do = D - dk + 1;
    size_t tot = (size_t)Do * P;
    for (size_t i = (size_t)blockIdx.x * blockDim.x + threadIdx.x; i < tot;
         i += (size_t)gridDim.x * blockDim.x) {
        int p = (int)(i % P); int z = (int)(i / P);
        unsigned char v = 0;
        for (int d = 0; d < dk; ++d) v |= m[(size_t)(z + d) * P + p];
        mo[i] = v;
    }
}

// --------------------------- masked BN + ReLU ------------------------------

__global__ void k_bn_stats(const float* __restrict__ x, const unsigned char* __restrict__ mask,
                           float* __restrict__ st, size_t nvox, int C) {
    int c = threadIdx.x;
    float s = 0.f, s2 = 0.f, cnt = 0.f;
    for (size_t v = blockIdx.x; v < nvox; v += gridDim.x) {
        float mk = (float)mask[v];
        float val = x[v * (size_t)C + c] * mk;
        s += val; s2 += val * val; cnt += mk;
    }
    atomicAdd(&st[c], s);
    atomicAdd(&st[C + c], s2);
    if (c == 0) atomicAdd(&st[2 * C], cnt);
}

__global__ void k_bn_apply(const float* __restrict__ x, const unsigned char* __restrict__ mask,
                           const float* __restrict__ g, const float* __restrict__ b,
                           const float* __restrict__ st,
                           unsigned short* __restrict__ act16, float* __restrict__ act32,
                           size_t nvox, int C, float eps) {
    size_t tot = nvox * (size_t)C;
    float cnt = st[2 * C]; cnt = cnt < 1.f ? 1.f : cnt;
    for (size_t i = (size_t)blockIdx.x * blockDim.x + threadIdx.x; i < tot;
         i += (size_t)gridDim.x * blockDim.x) {
        int c = (int)(i % C); size_t v = i / C;
        float mean = st[c] / cnt;
        float var  = st[C + c] / cnt - mean * mean;
        float y = (x[i] - mean) * rsqrtf(var + eps) * g[c] + b[c];
        y = fmaxf(y, 0.f) * (float)mask[v];
        if (act16) act16[i] = f2bf(y);
        if (act32) act32[i] = y;
    }
}

// --------------------------- WMMA implicit-GEMM conv -----------------------
// in  : bf16 activations [D,H,W,Cin]            (Cin multiple of 32)
// wgt : fragment-major bf16 weights (see k_swizzle_w)
// out : f32 [Do,Ho,Wo,Cout], masked by omask (may be null)
__global__ void __launch_bounds__(256)
k_conv3d_wmma(const unsigned short* __restrict__ in, const unsigned short* __restrict__ wgt,
              float* __restrict__ out, const unsigned char* __restrict__ omask,
              int D, int H, int W, int Cin,
              int Do, int Ho, int Wo, int Cout,
              int KD, int KH, int KW, int stride, int pad) {
    __shared__ __align__(16) unsigned short lds[14400];   // max KH*PW*(Cin+8)
    __shared__ int s_any;

    const int tilesW = (Wo + 15) >> 4;
    int t = blockIdx.x;
    const int tw = t % tilesW; t /= tilesW;
    const int oy = t % Ho;
    const int oz = t / Ho;
    const int x0 = tw << 4;

    // sparse fast path: whole tile masked out -> write zeros
    if (threadIdx.x == 0) {
        int any = (omask == nullptr);
        if (!any) {
            size_t base = ((size_t)oz * Ho + oy) * Wo;
            for (int i = 0; i < 16; ++i) {
                int ox = x0 + i;
                if (ox < Wo && omask[base + ox]) { any = 1; break; }
            }
        }
        s_any = any;
    }
    __syncthreads();
    if (!s_any) {
        int lim = (Wo - x0 < 16 ? Wo - x0 : 16) * Cout;
        size_t base = (((size_t)oz * Ho + oy) * Wo + x0) * Cout;
        for (int i = threadIdx.x; i < lim; i += blockDim.x) out[base + i] = 0.f;
        return;
    }

    const int PW = 15 * stride + KW;          // staged x extent
    const int Cpitch = Cin + 8;               // LDS row pitch (bank-conflict pad)
    const int iz0 = oz * stride - pad;
    const int iy0 = oy * stride - pad;
    const int ix0 = x0 * stride - pad;

    const int wave = (int)(threadIdx.x >> 5);
    const int lane = (int)(threadIdx.x & 31);
    const int m    = lane & 15;               // A row (voxel) / B col (cout)
    const int nh   = lane >> 4;               // lane-half selects K sub-block
    const int nSlices = Cout >> 4;
    const bool active = wave < nSlices;
    const size_t kqStride = (size_t)nSlices * 512;   // halves between K-chunks

#if USE_TDM
    const bool tdmOk = (ix0 >= 0) && (iy0 >= 0) &&
                       (Cin == 32 || Cin == 64 || Cin == 128 || Cin == 256);
    const int padIntCode = (Cin == 32) ? 3 : (Cin == 64) ? 4 : (Cin == 128) ? 5 : 6;
#endif

    v8f acc = {};

    for (int dz = 0; dz < KD; ++dz) {
        const int iz = iz0 + dz;
        bool staged = false;
#if USE_TDM
        if (tdmOk && iz >= 0 && iz < D) {
            if (threadIdx.x < 32) {
                // 2D TDM descriptor: KH rows of PW*Cin/2 dwords, row stride W*Cin/2,
                // tensor dims give free right/bottom OOB zero-fill, pad fields
                // reproduce the Cin+8 LDS pitch.
                unsigned ldsAddr = (unsigned)(size_t)&lds[0];
                size_t ga = (size_t)(const void*)(in + (((size_t)iz * H + iy0) * W + ix0) * (size_t)Cin);
                v4u_ g0v;
                g0v[0] = 1u;                                        // count=1
                g0v[1] = ldsAddr;                                   // lds_addr (bytes)
                g0v[2] = (unsigned)(ga & 0xFFFFFFFFu);              // global_addr[31:0]
                g0v[3] = (unsigned)((ga >> 32) & 0x1FFFFFFu) | (2u << 30);  // addr[56:32], type=2
                unsigned dim0  = (unsigned)((W - ix0) * (Cin >> 1));        // dwords in row
                unsigned dim1  = (unsigned)(H - iy0);                       // rows
                unsigned tile0 = (unsigned)(PW * (Cin >> 1));               // dwords per row
                unsigned long long str0 = (unsigned long long)W * (unsigned)(Cin >> 1);
                v8i_ g1v;
                g1v[0] = (int)((2u << 16) | (1u << 20) |                    // data_size=4B, pad_enable
                               ((unsigned)padIntCode << 22) | (3u << 25));  // interval=Cin/2 dw, pad=4 dw
                g1v[1] = (int)((dim0 & 0xFFFFu) << 16);
                g1v[2] = (int)((dim0 >> 16) | ((dim1 & 0xFFFFu) << 16));
                g1v[3] = (int)((dim1 >> 16) | (tile0 << 16));
                g1v[4] = (int)(unsigned)KH;                                 // tile_dim1, tile_dim2=0
                g1v[5] = (int)(unsigned)(str0 & 0xFFFFFFFFu);
                g1v[6] = (int)(unsigned)((str0 >> 32) & 0xFFFFu);
                g1v[7] = 0;
                v4i_ zz = {0, 0, 0, 0};
                v8i_ z8 = {0, 0, 0, 0, 0, 0, 0, 0};
                __builtin_amdgcn_tensor_load_to_lds(g0v, g1v, zz, zz, z8, 0);
                __builtin_amdgcn_s_wait_tensorcnt((short)0);
            }
            staged = true;
        }
#endif
        if (!staged) {
            const int planeElems = KH * PW * Cin;
            for (int i = (int)threadIdx.x; i < planeElems; i += (int)blockDim.x) {
                int c = i % Cin;
                int r = i / Cin;
                int px = r % PW;
                int dy = r / PW;
                int iy = iy0 + dy, ix = ix0 + px;
                unsigned short v = 0;
                if ((unsigned)iz < (unsigned)D && (unsigned)iy < (unsigned)H &&
                    (unsigned)ix < (unsigned)W)
                    v = in[(((size_t)iz * H + iy) * W + ix) * (size_t)Cin + c];
                lds[(size_t)r * Cpitch + c] = v;
            }
        }
        __syncthreads();
        if (active) {
            for (int dy = 0; dy < KH; ++dy) {
                const unsigned short* prow = &lds[(size_t)(dy * PW) * Cpitch];
                for (int dx = 0; dx < KW; ++dx) {
                    const int tap = (dz * KH + dy) * KW + dx;
                    const unsigned short* ap = &prow[(size_t)(m * stride + dx) * Cpitch];
                    const unsigned short* wfrag =
                        &wgt[(((size_t)tap * (Cin >> 5)) * nSlices + wave) * 512 + (size_t)lane * 16];
                    __builtin_prefetch((const void*)(wfrag + (size_t)(Cin >> 5) * kqStride), 0, 0);
                    #pragma unroll 2
                    for (int kc = 0; kc < Cin; kc += 32) {
                        WFrag A, B;
                        const uint4* bq = (const uint4*)wfrag;
                        B.q[0] = bq[0];                         // 2x global_load_b128
                        B.q[1] = bq[1];
                        A.q[0] = *(const uint4*)(ap + kc + nh * 8);        // ds_load_b128
                        A.q[1] = *(const uint4*)(ap + kc + 16 + nh * 8);   // ds_load_b128
                        acc = __builtin_amdgcn_wmma_f32_16x16x32_bf16(
                                  false, A.v, false, B.v, (short)0, acc, false, false);
                        wfrag += kqStride;
                    }
                }
            }
        }
        __syncthreads();
    }

    if (active) {
        const int coutBase = wave << 4;
        #pragma unroll
        for (int r = 0; r < 8; ++r) {
            int vox = r + nh * 8;             // f32 C/D layout: M = r + 8*(lane>=16)
            int ox = x0 + vox;
            if (ox < Wo) {
                size_t vbase = ((size_t)oz * Ho + oy) * Wo + ox;
                float val = acc[r];
                if (omask) val = omask[vbase] ? val : 0.f;
                out[vbase * Cout + coutBase + m] = val;
            }
        }
    }
}

// --------------------------- small VALU kernels ----------------------------

__global__ void k_conv_in(const unsigned short* __restrict__ g, const unsigned short* __restrict__ w,
                          float* __restrict__ out, const unsigned char* __restrict__ mask,
                          int G, int Cout) {
    size_t tot = (size_t)G * G * G * Cout;
    for (size_t i = (size_t)blockIdx.x * blockDim.x + threadIdx.x; i < tot;
         i += (size_t)gridDim.x * blockDim.x) {
        int co = (int)(i % Cout); size_t v = i / Cout;
        int x = (int)(v % G); int y = (int)((v / G) % G); int z = (int)(v / ((size_t)G * G));
        float a = 0.f;
        if (mask[v]) {
            for (int t = 0; t < 27; ++t) {
                int dx = t % 3, dy = (t / 3) % 3, dz = t / 9;
                int iz = z + dz - 1, iy = y + dy - 1, ix = x + dx - 1;
                if ((unsigned)iz < (unsigned)G && (unsigned)iy < (unsigned)G &&
                    (unsigned)ix < (unsigned)G)
                    a += bf2f(g[((size_t)iz * G + iy) * G + ix]) * bf2f(w[t * Cout + co]);
            }
        }
        out[i] = a;
    }
}

__global__ void k_deconv(const unsigned short* __restrict__ in, const unsigned short* __restrict__ w,
                         float* __restrict__ out, const unsigned char* __restrict__ omask,
                         int Di, int Do, int C) {
    size_t tot = (size_t)Do * Do * Do * C;
    for (size_t i = (size_t)blockIdx.x * blockDim.x + threadIdx.x; i < tot;
         i += (size_t)gridDim.x * blockDim.x) {
        int co = (int)(i % C); size_t v = i / C;
        int x = (int)(v % Do); int y = (int)((v / Do) % Do); int z = (int)(v / ((size_t)Do * Do));
        float a = 0.f;
        if (omask[v]) {
            for (int kd = 0; kd < 3; ++kd) {
                int tz = z - kd; if (tz < 0 || (tz & 1)) continue; int iz = tz >> 1; if (iz >= Di) continue;
                for (int kh = 0; kh < 3; ++kh) {
                    int ty = y - kh; if (ty < 0 || (ty & 1)) continue; int iy = ty >> 1; if (iy >= Di) continue;
                    for (int kw = 0; kw < 3; ++kw) {
                        int tx = x - kw; if (tx < 0 || (tx & 1)) continue; int ix = tx >> 1; if (ix >= Di) continue;
                        const unsigned short* ip = &in[(((size_t)iz * Di + iy) * Di + ix) * C];
                        const unsigned short* wp = &w[(size_t)((kd * 3 + kh) * 3 + kw) * C * C + co];
                        float s = 0.f;
                        for (int c = 0; c < C; ++c) s += bf2f(ip[c]) * bf2f(wp[(size_t)c * C]);
                        a += s;
                    }
                }
            }
        }
        out[i] = a;
    }
}

__global__ void k_bev(const float* __restrict__ x, float* __restrict__ dst, int Do, int L, int C) {
    size_t tot = (size_t)Do * L * L * C;
    for (size_t i = (size_t)blockIdx.x * blockDim.x + threadIdx.x; i < tot;
         i += (size_t)gridDim.x * blockDim.x) {
        int c = (int)(i % C); size_t v = i / C;
        int xx = (int)(v % L); int yy = (int)((v / L) % L); int z = (int)(v / ((size_t)L * L));
        dst[(((size_t)c * Do + z) * L + yy) * L + xx] = x[i];
    }
}

// --------------------------------- host ------------------------------------

static inline unsigned blocks_for(size_t n) {
    size_t b = (n + 255) / 256;
    if (b == 0) b = 1;
    if (b > 1048576) b = 1048576;
    return (unsigned)b;
}

extern "C" void kernel_launch(void* const* d_in, const int* in_sizes, int n_in,
                              void* d_out, int out_size, void* d_ws, size_t ws_size,
                              hipStream_t stream) {
    (void)out_size; (void)ws_size;
    if (n_in < 171) return;   // 3 fixed inputs + 168 param leaves (jax tree order)

    const float* vf    = (const float*)d_in[0];
    const int*   coors = (const int*)d_in[1];
    const int    nVoxIn = in_sizes[0];

    // ---- bump allocator over d_ws ----
    char* wsp = (char*)d_ws;
    size_t off = 0;
    auto alloc = [&](size_t bytes) -> void* {
        void* p = wsp + off;
        off = (off + bytes + 255) & ~(size_t)255;
        return p;
    };
    auto tobf = [&](const float* src, size_t n) -> unsigned short* {
        unsigned short* d = (unsigned short*)alloc(n * 2);
        k_f32_to_bf16<<<blocks_for(n), 256, 0, stream>>>(src, d, n);
        return d;
    };
    auto toswz = [&](const float* src, int taps, int Cin, int Cout) -> unsigned short* {
        size_t n = (size_t)taps * Cin * Cout;
        unsigned short* d = (unsigned short*)alloc(n * 2);
        k_swizzle_w<<<blocks_for(n), 256, 0, stream>>>(src, d, taps, Cin, Cout);
        return d;
    };

    // ---- parameter leaves: jax.tree_util order = dicts sorted by key ----
    int cur = 3;
    auto nxt = [&]() { return (const float*)d_in[cur++]; };
    auto nxtW = [&]() { size_t n = (size_t)in_sizes[cur]; const float* p = nxt(); return tobf(p, n); };

    struct RBp { const float *g1, *b1, *g2, *b2; unsigned short *w1, *w2, *nin; int a, b; };
    auto readRB = [&](int a, int b) {
        RBp r; r.a = a; r.b = b;
        r.b1 = nxt(); r.b2 = nxt(); r.g1 = nxt(); r.g2 = nxt();   // 'b1','b2','g1','g2'
        r.nin = (a != b) ? toswz(nxt(), 1, a, b) : nullptr;       // 'nin'
        r.w1 = toswz(nxt(), 27, a, b);                            // 'w1'
        r.w2 = toswz(nxt(), 27, b, b);                            // 'w2'
        return r;
    };
    // 'cat0','cat1','cat2'
    RBp cat0b[2] = { readRB(256, 128), readRB(128, 128) };
    RBp cat1b[2] = { readRB(256, 128), readRB(128, 128) };
    RBp cat2b[2] = { readRB(256, 128), readRB(128, 128) };
    // 'down' : list of {'b','blocks','g','w'}
    struct DownP { const float *g, *b; unsigned short* w; RBp blk[2]; };
    auto readDown = [&](int a, int bch) {
        DownP d; d.b = nxt(); d.blk[0] = readRB(a, a); d.blk[1] = readRB(a, a);
        d.g = nxt(); d.w = toswz(nxt(), 27, a, bch); return d;
    };
    DownP down0 = readDown(32, 64), down1 = readDown(64, 96), down2 = readDown(96, 128);
    // 'fm0','fm1','fm2' : {'b1','b2','g1','g2','w'};  'fm3' : {'b','g'}
    struct FMp { const float *g1, *b1, *g2, *b2; unsigned short* w; };
    auto readFM = [&](int dk, int co) {
        FMp f; f.b1 = nxt(); f.b2 = nxt(); f.g1 = nxt(); f.g2 = nxt();
        f.w = toswz(nxt(), dk, 128, co); return f;
    };
    FMp fm0 = readFM(15, 32), fm1 = readFM(7, 64), fm2 = readFM(3, 96);
    const float* fm3_b = nxt(); const float* fm3_g = nxt();
    // 'up10','up21','up32' : {'b','g','w'}   (deconv keeps plain layout)
    struct UPp { const float *g, *b; unsigned short* w; };
    auto readUP = [&]() { UPp u; u.b = nxt(); u.g = nxt(); u.w = nxtW(); return u; };
    UPp up10 = readUP(), up21 = readUP(), up32 = readUP();
    // 'w_in'  (stem keeps plain layout)
    unsigned short* w_in_b = nxtW();
    // 'x0' : 2 RBs;  'x1'..'x3' : {'b','blocks','g','w'}
    RBp x0b[2] = { readRB(128, 128), readRB(128, 128) };
    struct XKp { const float *g, *b; unsigned short* w; RBp blk[2]; };
    auto readXK = [&]() {
        XKp k; k.b = nxt(); k.blk[0] = readRB(128, 128); k.blk[1] = readRB(128, 128);
        k.g = nxt(); k.w = toswz(nxt(), 27, 128, 128); return k;
    };
    XKp x1p = readXK(), x2p = readXK(), x3p = readXK();

    // ---- grid levels ----
    const int Ls[7] = {127, 63, 31, 15, 7, 3, 1};
    size_t NV[7];
    for (int i = 0; i < 7; ++i) NV[i] = (size_t)Ls[i] * Ls[i] * Ls[i];

    unsigned char* mk[7];
    for (int i = 0; i < 7; ++i) mk[i] = (unsigned char*)alloc(NV[i]);

    unsigned short* g0 = (unsigned short*)alloc(NV[0] * 2);
    float* stats = (float*)alloc(4096);
    unsigned short* sA = (unsigned short*)alloc(NV[0] * 32 * 2);
    float* sT1 = (float*)alloc(NV[0] * 32 * 4);
    float* sT2 = (float*)alloc(NV[0] * 32 * 4);

    // ---- scatter + masks ----
    k_zero_u16<<<blocks_for(NV[0]), 256, 0, stream>>>(g0, NV[0]);
    k_zero_u8<<<blocks_for(NV[0]), 256, 0, stream>>>(mk[0], NV[0]);
    k_scatter<<<blocks_for((size_t)nVoxIn), 256, 0, stream>>>(vf, coors, g0, mk[0], nVoxIn, 127);
    for (int l = 1; l < 7; ++l)
        k_down_mask<<<blocks_for(NV[l]), 256, 0, stream>>>(mk[l - 1], mk[l], Ls[l - 1], Ls[l]);

    // ---- helpers ----
    auto bnrelu = [&](const float* x, const unsigned char* mask, const float* g, const float* b,
                      unsigned short* act16, float* act32, size_t nvox, int C, float eps) {
        k_zero_f32<<<1, 256, 0, stream>>>(stats, (size_t)(2 * C + 1));
        unsigned gb = (unsigned)(nvox < 2048 ? nvox : 2048);
        k_bn_stats<<<gb, C, 0, stream>>>(x, mask, stats, nvox, C);
        k_bn_apply<<<blocks_for(nvox * (size_t)C), 256, 0, stream>>>(
            x, mask, g, b, stats, act16, act32, nvox, C, eps);
    };
    auto conv = [&](const unsigned short* act, const unsigned short* w, float* out,
                    const unsigned char* omask, int D, int H, int W, int Do, int Ho, int Wo,
                    int Cin, int Cout, int KD, int KH, int KW, int stride, int pad) {
        int tilesW = (Wo + 15) / 16;
        int waves = Cout / 16; if (waves > 8) waves = 8;
        k_conv3d_wmma<<<dim3((unsigned)(tilesW * Ho * Do)), dim3(32 * waves), 0, stream>>>(
            act, w, out, omask, D, H, W, Cin, Do, Ho, Wo, Cout, KD, KH, KW, stride, pad);
    };
    auto resblock = [&](const float* src, float* dst, const unsigned char* mask,
                        const RBp& p, int L) {
        size_t nv = (size_t)L * L * L;
        bnrelu(src, mask, p.g1, p.b1, sA, nullptr, nv, p.a, 1e-4f);
        conv(sA, p.w1, sT1, mask, L, L, L, L, L, L, p.a, p.b, 3, 3, 3, 1, 1);
        bnrelu(sT1, mask, p.g2, p.b2, sA, nullptr, nv, p.b, 1e-4f);
        conv(sA, p.w2, sT2, mask, L, L, L, L, L, L, p.b, p.b, 3, 3, 3, 1, 1);
        if (p.nin) {
            k_f32_to_bf16<<<blocks_for(nv * p.a), 256, 0, stream>>>(src, sA, nv * (size_t)p.a);
            conv(sA, p.nin, sT1, mask, L, L, L, L, L, L, p.a, p.b, 1, 1, 1, 1, 0);
            k_add2_f32<<<blocks_for(nv * p.b), 256, 0, stream>>>(dst, sT1, sT2, nv * (size_t)p.b);
        } else {
            k_add2_f32<<<blocks_for(nv * p.b), 256, 0, stream>>>(dst, src, sT2, nv * (size_t)p.b);
        }
    };

    // ---- stem + encoder ----
    float* h0 = (float*)alloc(NV[0] * 32 * 4);
    k_conv_in<<<blocks_for(NV[0] * 32), 256, 0, stream>>>(g0, w_in_b, h0, mk[0], 127, 32);

    float* hcur = h0; int Cc = 32;
    DownP* dps[3] = { &down0, &down1, &down2 };
    const int Cn[3] = {64, 96, 128};
    for (int l = 0; l < 3; ++l) {
        resblock(hcur, hcur, mk[l], dps[l]->blk[0], Ls[l]);
        resblock(hcur, hcur, mk[l], dps[l]->blk[1], Ls[l]);
        bnrelu(hcur, mk[l], dps[l]->g, dps[l]->b, sA, nullptr, NV[l], Cc, 1e-4f);
        float* hn = (float*)alloc(NV[l + 1] * Cn[l] * 4);
        conv(sA, dps[l]->w, hn, mk[l + 1], Ls[l], Ls[l], Ls[l],
             Ls[l + 1], Ls[l + 1], Ls[l + 1], Cc, Cn[l], 3, 3, 3, 2, 0);
        hcur = hn; Cc = Cn[l];
    }
    resblock(hcur, hcur, mk[3], x0b[0], 15);
    resblock(hcur, hcur, mk[3], x0b[1], 15);

    float* X[4]; X[0] = hcur;
    XKp* xps[3] = { &x1p, &x2p, &x3p };
    for (int k = 1; k <= 3; ++k) {
        XKp* p = xps[k - 1];
        bnrelu(X[k - 1], mk[2 + k], p->g, p->b, sA, nullptr, NV[2 + k], 128, 1e-4f);
        X[k] = (float*)alloc(NV[3 + k] * 128 * 4);
        conv(sA, p->w, X[k], mk[3 + k], Ls[2 + k], Ls[2 + k], Ls[2 + k],
             Ls[3 + k], Ls[3 + k], Ls[3 + k], 128, 128, 3, 3, 3, 2, 0);
        resblock(X[k], X[k], mk[3 + k], p->blk[0], Ls[3 + k]);
        resblock(X[k], X[k], mk[3 + k], p->blk[1], Ls[3 + k]);
    }

    // ---- decoder ----
    auto upfn = [&](const float* xx, const unsigned char* mi, UPp& p, int Li,
                    const unsigned char* mo_, int Lo, float* out) {
        size_t nvi = (size_t)Li * Li * Li, nvo = (size_t)Lo * Lo * Lo;
        bnrelu(xx, mi, p.g, p.b, sA, nullptr, nvi, 128, 1e-4f);
        k_deconv<<<blocks_for(nvo * 128), 256, 0, stream>>>(sA, p.w, out, mo_, Li, Lo, 128);
    };
    float* cbuf = (float*)alloc(NV[3] * 256 * 4);

    float* u2  = (float*)alloc(NV[5] * 128 * 4);
    float* x2f = (float*)alloc(NV[5] * 128 * 4);
    upfn(X[3], mk[6], up32, 1, mk[5], 3, u2);
    k_concat<<<blocks_for(NV[5] * 256), 256, 0, stream>>>(X[2], u2, cbuf, NV[5], 128, 128);
    resblock(cbuf, x2f, mk[5], cat2b[0], 3);
    resblock(x2f, x2f, mk[5], cat2b[1], 3);

    float* u1  = (float*)alloc(NV[4] * 128 * 4);
    float* x1f = (float*)alloc(NV[4] * 128 * 4);
    upfn(x2f, mk[5], up21, 3, mk[4], 7, u1);
    k_concat<<<blocks_for(NV[4] * 256), 256, 0, stream>>>(X[1], u1, cbuf, NV[4], 128, 128);
    resblock(cbuf, x1f, mk[4], cat1b[0], 7);
    resblock(x1f, x1f, mk[4], cat1b[1], 7);

    float* u0  = (float*)alloc(NV[3] * 128 * 4);
    float* x0f = (float*)alloc(NV[3] * 128 * 4);
    upfn(x1f, mk[4], up10, 7, mk[3], 15, u0);
    k_concat<<<blocks_for(NV[3] * 256), 256, 0, stream>>>(X[0], u0, cbuf, NV[3], 128, 128);
    resblock(cbuf, x0f, mk[3], cat0b[0], 15);
    resblock(x0f, x0f, mk[3], cat0b[1], 15);

    // ---- feature maps -> BEV outputs ----
    float* outf = (float*)d_out;
    auto feat = [&](const float* x, const unsigned char* m, FMp& p, int L, int dk, int co, float* dst) {
        size_t nv = (size_t)L * L * L;
        bnrelu(x, m, p.g1, p.b1, sA, nullptr, nv, 128, 1e-4f);
        int Do = L - dk + 1;
        conv(sA, p.w, sT1, nullptr, L, L, L, Do, L, L, 128, co, dk, 1, 1, 1, 0);
        unsigned char* m2 = (unsigned char*)alloc((size_t)Do * L * L);
        k_reduce_mask_d<<<blocks_for((size_t)Do * L * L), 256, 0, stream>>>(m, m2, L, L * L, dk);
        bnrelu(sT1, m2, p.g2, p.b2, nullptr, sT2, (size_t)Do * L * L, co, 1e-3f);
        k_bev<<<blocks_for((size_t)Do * L * L * co), 256, 0, stream>>>(sT2, dst, Do, L, co);
    };
    feat(x0f, mk[3], fm0, 15, 15, 32, outf);                       // (32,15,15)
    feat(x1f, mk[4], fm1, 7, 7, 64, outf + 7200);                  // (64,7,7)
    feat(x2f, mk[5], fm2, 3, 3, 96, outf + 7200 + 3136);           // (96,3,3)
    bnrelu(X[3], mk[6], fm3_g, fm3_b, nullptr, sT2, 1, 128, 1e-4f);
    k_bev<<<1, 256, 0, stream>>>(sT2, outf + 7200 + 3136 + 864, 1, 1, 128);  // (128,1,1)
}
